// Qwen3TTSTalkerAttentionStandalone_17583596109939
// MI455X (gfx1250) — compile-verified, matrix operand order checked by reading, not run on hardware
//
#include <hip/hip_runtime.h>
#include <hip/hip_bf16.h>

// ---------------- constants for this problem ----------------
#define HD   128
#define NH   16
#define NKV  8
#define BB   2
#define SS   2048
#define DD   2048
#define MS   4096          // B*S rows

typedef __attribute__((ext_vector_type(16))) __bf16 v16bf;
typedef __attribute__((ext_vector_type(8)))  float  v8f;
typedef __attribute__((ext_vector_type(8)))  __bf16 bf16x8;
typedef __attribute__((ext_vector_type(4)))  __bf16 bf16x4;

__device__ __forceinline__ __bf16 f2bf(float f) {
  unsigned u = __float_as_uint(f);
  u += 0x7FFFu + ((u >> 16) & 1u);          // round-to-nearest-even
  unsigned short h = (unsigned short)(u >> 16);
  __bf16 r;
  __builtin_memcpy(&r, &h, 2);
  return r;
}

// LDS byte address (workgroup-relative) from a generic pointer:
// per CDNA5 aperture rules the LDS address is addr[31:0].
__device__ __forceinline__ unsigned lds_a32(const void* p) {
  return (unsigned)(unsigned long long)(uintptr_t)p;
}

// Async DMA: global memory -> LDS, 16B per lane, tracked by ASYNCcnt.
__device__ __forceinline__ void async_ld_b128(unsigned lds_off, const void* g) {
  asm volatile("global_load_async_to_lds_b128 %0, %1, off"
               :: "v"(lds_off), "v"((unsigned long long)(uintptr_t)g)
               : "memory");
}

// ---------------- f32 -> bf16 conversion ----------------
__global__ void cvt_f32_bf16(const float* __restrict__ src,
                             __bf16* __restrict__ dst, int n) {
  int i = blockIdx.x * blockDim.x + threadIdx.x;
  int stride = gridDim.x * blockDim.x;
  for (; i < n; i += stride) dst[i] = f2bf(src[i]);
}

// ---------------- tiled transpose + convert: f32 [R,C] -> bf16 [C,R] ----------
__global__ __launch_bounds__(256)
void trans_f32_bf16(const float* __restrict__ src, __bf16* __restrict__ dst,
                    int R, int C) {
  __shared__ float tl[32][33];
  const int tx = threadIdx.x & 31, ty = threadIdx.x >> 5;
  const int r0 = blockIdx.y * 32, c0 = blockIdx.x * 32;
#pragma unroll
  for (int i = 0; i < 32; i += 8)
    tl[ty + i][tx] = src[(size_t)(r0 + ty + i) * C + c0 + tx];
  __syncthreads();
#pragma unroll
  for (int i = 0; i < 32; i += 8)
    dst[(size_t)(c0 + ty + i) * R + r0 + tx] = f2bf(tl[tx][ty + i]);
}

// ---------------- batched tiled transpose: bf16 [z][R,C] -> bf16 [z][C,R] ----
__global__ __launch_bounds__(256)
void trans_bf16(const __bf16* __restrict__ src, __bf16* __restrict__ dst,
                int R, int C) {
  __shared__ __bf16 tl[32][34];
  const size_t base = (size_t)blockIdx.z * R * C;
  const int tx = threadIdx.x & 31, ty = threadIdx.x >> 5;
  const int r0 = blockIdx.y * 32, c0 = blockIdx.x * 32;
#pragma unroll
  for (int i = 0; i < 32; i += 8)
    tl[ty + i][tx] = src[base + (size_t)(r0 + ty + i) * C + c0 + tx];
  __syncthreads();
#pragma unroll
  for (int i = 0; i < 32; i += 8)
    dst[base + (size_t)(c0 + ty + i) * R + r0 + tx] = tl[tx][ty + i];
}

// ---------------- mRoPE cos/sin merge ----------------
// sections [16,24,24]*2 -> modality 0,1,2,0,1,2 over head dim
__device__ __forceinline__ int mrope_mod(int d) {
  int dd = (d < 64) ? d : d - 64;
  return dd < 16 ? 0 : (dd < 40 ? 1 : 2);
}
__global__ void mrope_build(const float* __restrict__ cos3,
                            const float* __restrict__ sin3,
                            float* __restrict__ c, float* __restrict__ s) {
  int i = blockIdx.x * blockDim.x + threadIdx.x;   // over B*S*HD
  if (i >= BB * SS * HD) return;
  int d = i & (HD - 1);
  size_t src = (size_t)mrope_mod(d) * BB * SS * HD + (size_t)i;
  c[i] = cos3[src];
  s[i] = sin3[src];
}

// ---------------- generic bf16 WMMA GEMM ----------------
// C[M,N] f32 = A[M,K] bf16 (row major) * Wt[N,K] bf16 (row major = W^T)
// 256x128 block tile, 256 threads = 8 waves, each wave 64x64 (4x4 WMMA accs).
// Both tiles are row-major copies -> async DMA, double-buffered LDS.
// Per k-step per wave: 4 B-frags preloaded + 4 A-frags -> 16 ds_load : 16 wmma.
#define GT_M 256
#define GT_N 128
#define GT_K 32
#define LDT  40   // padded LDS row stride (elements); 80B keeps 16B alignment

__global__ __launch_bounds__(256)
void gemm_bf16(const __bf16* __restrict__ A, const __bf16* __restrict__ Wt,
               float* __restrict__ C, int M, int N, int K) {
  __shared__ __bf16 As[2][GT_M * LDT];   // [row][k]
  __shared__ __bf16 Bs[2][GT_N * LDT];   // [n][k] (Wt rows)
  const int t = threadIdx.x;
  const int lane = t & 31, wv = t >> 5;
  const int wm = wv & 3, wn = wv >> 2;          // wave tile origin: 64x64
  const int m0 = blockIdx.y * GT_M, n0 = blockIdx.x * GT_N;
  const int lm = lane & 15, kh = lane >> 4;

  // tile copies in b128 (8 bf16) chunks:
  // A: 256x32 = 1024 chunks -> 4/thread; B: 128x32 = 512 chunks -> 2/thread
  const int crow = t >> 2;               // 0..63
  const int ccol = (t & 3) * 8;          // elem chunk 0,8,16,24
  const __bf16* ga[4];
#pragma unroll
  for (int i = 0; i < 4; ++i)
    ga[i] = A + (size_t)(m0 + crow + 64 * i) * K + ccol;
  const __bf16* gb0 = Wt + (size_t)(n0 + crow) * K + ccol;
  const __bf16* gb1 = Wt + (size_t)(n0 + crow + 64) * K + ccol;

  auto issue = [&](int k0, int buf) {
#pragma unroll
    for (int i = 0; i < 4; ++i)
      async_ld_b128(lds_a32(&As[buf][(crow + 64 * i) * LDT + ccol]), ga[i] + k0);
    async_ld_b128(lds_a32(&Bs[buf][crow * LDT + ccol]),        gb0 + k0);
    async_ld_b128(lds_a32(&Bs[buf][(crow + 64) * LDT + ccol]), gb1 + k0);
  };

  v8f acc[4][4];
#pragma unroll
  for (int mt = 0; mt < 4; ++mt)
#pragma unroll
    for (int nt = 0; nt < 4; ++nt)
#pragma unroll
      for (int j = 0; j < 8; ++j) acc[mt][nt][j] = 0.f;

  const int T = K / GT_K;
  issue(0, 0);
  for (int kt = 0; kt < T; ++kt) {
    if (kt + 1 < T) {
      issue((kt + 1) * GT_K, (kt + 1) & 1);       // overlap next tile's DMA
      asm volatile("s_wait_asynccnt 0x6" ::: "memory");  // this tile landed
    } else {
      asm volatile("s_wait_asynccnt 0x0" ::: "memory");
    }
    if (kt + 2 < T) __builtin_prefetch(ga[0] + (kt + 2) * GT_K, 0, 0);
    __syncthreads();

    const __bf16* as = As[kt & 1];
    const __bf16* bs = Bs[kt & 1];

    // preload 4 B-fragments (reused across all 4 A-fragments)
    v16bf bfr[4];
#pragma unroll
    for (int nt = 0; nt < 4; ++nt) {
      int n = wn * 64 + nt * 16 + lm;
      bf16x8 blo = *(const bf16x8*)&bs[n * LDT + kh * 8];
      bf16x8 bhi = *(const bf16x8*)&bs[n * LDT + 16 + kh * 8];
#pragma unroll
      for (int j = 0; j < 8; ++j) { bfr[nt][j] = blo[j]; bfr[nt][8 + j] = bhi[j]; }
    }
#pragma unroll
    for (int mt = 0; mt < 4; ++mt) {
      // A fragment (16x32 bf16): lane lm = row, kh selects K half-groups
      int m = wm * 64 + mt * 16 + lm;
      bf16x8 alo = *(const bf16x8*)&as[m * LDT + kh * 8];
      bf16x8 ahi = *(const bf16x8*)&as[m * LDT + 16 + kh * 8];
      v16bf af;
#pragma unroll
      for (int j = 0; j < 8; ++j) { af[j] = alo[j]; af[8 + j] = ahi[j]; }
#pragma unroll
      for (int nt = 0; nt < 4; ++nt)
        acc[mt][nt] = __builtin_amdgcn_wmma_f32_16x16x32_bf16(
            false, af, false, bfr[nt], (short)0, acc[mt][nt], false, false);
    }
    __syncthreads();
  }

  // C/D layout: VGPR r -> row r + 8*(lane>>4); col = lane&15
#pragma unroll
  for (int mt = 0; mt < 4; ++mt)
#pragma unroll
    for (int nt = 0; nt < 4; ++nt)
#pragma unroll
      for (int r = 0; r < 8; ++r) {
        int m = m0 + wm * 64 + mt * 16 + r + 8 * kh;
        int n = n0 + wn * 64 + nt * 16 + lm;
        C[(size_t)m * N + n] = acc[mt][nt][r];
      }
}

// ---------------- RMSNorm + mRoPE + head transpose ----------------
// src: f32 [B*S, HH, 128]  ->  dst: bf16 [B, HH, S, 128]
__global__ __launch_bounds__(128)
void norm_rope(const float* __restrict__ src, const float* __restrict__ w,
               const float* __restrict__ c, const float* __restrict__ s,
               __bf16* __restrict__ dst, int HH, int do_rope, int total) {
  const int lane = threadIdx.x & 31;
  const int wid = blockIdx.x * (blockDim.x >> 5) + (threadIdx.x >> 5);
  if (wid >= total) return;
  const int h = wid % HH;
  const int bs = wid / HH;
  const int sq = bs % SS, b = bs / SS;

  const float* xp = src + ((size_t)bs * HH + h) * HD;
  const int d = lane * 4;
  const int dp = (d + 64) & (HD - 1);
  float4 x  = *(const float4*)(xp + d);
  float4 xq = *(const float4*)(xp + dp);

  float ss = x.x * x.x + x.y * x.y + x.z * x.z + x.w * x.w;
#pragma unroll
  for (int off = 16; off; off >>= 1) ss += __shfl_xor(ss, off, 32);

  float inv = 1.0f;
  float4 w0 = make_float4(1.f, 1.f, 1.f, 1.f), w1 = w0;
  if (w) {
    inv = rsqrtf(ss * (1.0f / HD) + 1e-6f);
    w0 = *(const float4*)(w + d);
    w1 = *(const float4*)(w + dp);
  }
  float y[4]  = {x.x * w0.x * inv, x.y * w0.y * inv, x.z * w0.z * inv, x.w * w0.w * inv};
  float yp[4] = {xq.x * w1.x * inv, xq.y * w1.y * inv, xq.z * w1.z * inv, xq.w * w1.w * inv};

  float o[4];
  if (do_rope) {
    float4 cv = *(const float4*)(c + (size_t)bs * HD + d);
    float4 sv = *(const float4*)(s + (size_t)bs * HD + d);
    float sign = (d < 64) ? -1.f : 1.f;   // rotate_half
    o[0] = y[0] * cv.x + sign * yp[0] * sv.x;
    o[1] = y[1] * cv.y + sign * yp[1] * sv.y;
    o[2] = y[2] * cv.z + sign * yp[2] * sv.z;
    o[3] = y[3] * cv.w + sign * yp[3] * sv.w;
  } else {
#pragma unroll
    for (int j = 0; j < 4; ++j) o[j] = y[j];
  }
  bf16x4 ov;
#pragma unroll
  for (int j = 0; j < 4; ++j) ov[j] = f2bf(o[j]);
  *(bf16x4*)&dst[(((size_t)b * HH + h) * SS + sq) * HD + d] = ov;
}

// ---------------- flash attention (causal, GQA) ----------------
// One block: 64 q rows of one (b,h). 4 waves, each wave = 16 q rows.
// K tiles [key][dim] and pre-transposed V tiles [dim][key] are both row-major
// copies -> async DMA, double-buffered.
#define KSTR 136   // Ks row stride (elems): 272B, 16B aligned
#define VTS  40    // Vt row stride
#define PSTR 40    // P staging row stride

__global__ __launch_bounds__(128)
void attn_fa(const __bf16* __restrict__ Q, const __bf16* __restrict__ Kh,
             const __bf16* __restrict__ VhT, __bf16* __restrict__ O) {
  __shared__ __bf16 Ks[2][32 * KSTR];    // [key][dim]
  __shared__ __bf16 Vs[2][HD * VTS];     // [dim][key]
  __shared__ __bf16 Ps[4 * 16 * PSTR];

  const int t = threadIdx.x, lane = t & 31, wv = t >> 5;
  const int lm = lane & 15, kh = lane >> 4;
  const int q0 = blockIdx.x * 64;
  const int h = blockIdx.y, b = blockIdx.z, kv = h >> 1;
  const float scale = 0.08838834764831845f;   // hd^-0.5

  // Q fragments for this wave's 16 rows (A-layout, 4 K-chunks of 32)
  const int qrow = q0 + wv * 16 + lm;
  const __bf16* qptr = Q + ((size_t)(b * NH + h) * SS + qrow) * HD;
  v16bf qa[4];
#pragma unroll
  for (int kc = 0; kc < 4; ++kc) {
    bf16x8 lo = *(const bf16x8*)&qptr[kc * 32 + kh * 8];
    bf16x8 hi = *(const bf16x8*)&qptr[kc * 32 + 16 + kh * 8];
#pragma unroll
    for (int j = 0; j < 8; ++j) { qa[kc][j] = lo[j]; qa[kc][8 + j] = hi[j]; }
  }

  float mi[8], li[8];
  v8f ao[8];
#pragma unroll
  for (int r = 0; r < 8; ++r) { mi[r] = -1e30f; li[r] = 0.f; }
#pragma unroll
  for (int nt = 0; nt < 8; ++nt)
#pragma unroll
    for (int j = 0; j < 8; ++j) ao[nt][j] = 0.f;

  const __bf16* kbase  = Kh  + (size_t)(b * NKV + kv) * SS * HD;  // [S][hd]
  const __bf16* vbaseT = VhT + (size_t)(b * NKV + kv) * HD * SS;  // [hd][S]
  const int ntiles = (q0 + 64) / 32;            // causal upper bound

  // K tile: 32x128 = 512 b128 chunks / 128 threads = 4 each
  const int kr  = t >> 4;            // 0..7 (+ i*8)
  const int kc8 = (t & 15) * 8;      // elem chunk
  auto issueKV = [&](int kt, int buf) {
#pragma unroll
    for (int i = 0; i < 4; ++i) {
      int row = kr + i * 8;
      async_ld_b128(lds_a32(&Ks[buf][row * KSTR + kc8]),
                    &kbase[(size_t)(kt * 32 + row) * HD + kc8]);
    }
    // V tile: rows = 128 dims (one per thread), 4 chunks of 8 keys
    const __bf16* vb = vbaseT + (size_t)t * SS + kt * 32;
#pragma unroll
    for (int c2 = 0; c2 < 4; ++c2)
      async_ld_b128(lds_a32(&Vs[buf][t * VTS + c2 * 8]), vb + c2 * 8);
  };

  issueKV(0, 0);
  for (int kt = 0; kt < ntiles; ++kt) {
    if (kt + 1 < ntiles) {
      issueKV(kt + 1, (kt + 1) & 1);            // overlap next tile's DMA
      asm volatile("s_wait_asynccnt 0x8" ::: "memory");  // 8 in flight max
    } else {
      asm volatile("s_wait_asynccnt 0x0" ::: "memory");
    }
    __syncthreads();
    const __bf16* ks = Ks[kt & 1];
    const __bf16* vs = Vs[kt & 1];

    // scores: Q(16x128) x K^T -> 16x32 (two 16x16 C tiles)
    v8f sc[2];
#pragma unroll
    for (int nt = 0; nt < 2; ++nt) {
#pragma unroll
      for (int j = 0; j < 8; ++j) sc[nt][j] = 0.f;
      int n = nt * 16 + lm;
#pragma unroll
      for (int kc = 0; kc < 4; ++kc) {
        bf16x8 lo = *(const bf16x8*)&ks[n * KSTR + kc * 32 + kh * 8];
        bf16x8 hi = *(const bf16x8*)&ks[n * KSTR + kc * 32 + 16 + kh * 8];
        v16bf kf;
#pragma unroll
        for (int j = 0; j < 8; ++j) { kf[j] = lo[j]; kf[8 + j] = hi[j]; }
        sc[nt] = __builtin_amdgcn_wmma_f32_16x16x32_bf16(
            false, qa[kc], false, kf, (short)0, sc[nt], false, false);
      }
    }

    // scale + causal mask (element (r): row = r+8*kh, col = lane&15)
    const int qg = q0 + wv * 16;
#pragma unroll
    for (int nt = 0; nt < 2; ++nt)
#pragma unroll
      for (int r = 0; r < 8; ++r) {
        int qq = qg + r + 8 * kh;
        int kk = kt * 32 + nt * 16 + lm;
        float x = sc[nt][r] * scale;
        sc[nt][r] = (kk <= qq) ? x : -1e30f;
      }

    // online softmax: row reductions across the 16-lane half
    float tmax[8], tsum[8], alpha[8];
#pragma unroll
    for (int r = 0; r < 8; ++r) tmax[r] = fmaxf(sc[0][r], sc[1][r]);
#pragma unroll
    for (int off = 1; off < 16; off <<= 1)
#pragma unroll
      for (int r = 0; r < 8; ++r)
        tmax[r] = fmaxf(tmax[r], __shfl_xor(tmax[r], off, 32));
#pragma unroll
    for (int r = 0; r < 8; ++r) {
      float mn = fmaxf(mi[r], tmax[r]);
      alpha[r] = __expf(mi[r] - mn);
      mi[r] = mn;
    }
#pragma unroll
    for (int nt = 0; nt < 2; ++nt)
#pragma unroll
      for (int r = 0; r < 8; ++r) sc[nt][r] = __expf(sc[nt][r] - mi[r]);
#pragma unroll
    for (int r = 0; r < 8; ++r) tsum[r] = sc[0][r] + sc[1][r];
#pragma unroll
    for (int off = 1; off < 16; off <<= 1)
#pragma unroll
      for (int r = 0; r < 8; ++r) tsum[r] += __shfl_xor(tsum[r], off, 32);
#pragma unroll
    for (int r = 0; r < 8; ++r) li[r] = li[r] * alpha[r] + tsum[r];
#pragma unroll
    for (int nt = 0; nt < 8; ++nt)
#pragma unroll
      for (int r = 0; r < 8; ++r) ao[nt][r] *= alpha[r];

    // re-layout P (C-tile -> A-fragment) via per-wave LDS staging
    __bf16* pw = &Ps[wv * 16 * PSTR];
#pragma unroll
    for (int nt = 0; nt < 2; ++nt)
#pragma unroll
      for (int r = 0; r < 8; ++r)
        pw[(r + 8 * kh) * PSTR + nt * 16 + lm] = f2bf(sc[nt][r]);
    asm volatile("s_wait_dscnt 0" ::: "memory");   // wave-local RAW on LDS
    v16bf pf;
    {
      bf16x8 lo = *(const bf16x8*)&pw[lm * PSTR + kh * 8];
      bf16x8 hi = *(const bf16x8*)&pw[lm * PSTR + 16 + kh * 8];
#pragma unroll
      for (int j = 0; j < 8; ++j) { pf[j] = lo[j]; pf[8 + j] = hi[j]; }
    }

    // O += P(16x32) x V(32x128)
#pragma unroll
    for (int nt = 0; nt < 8; ++nt) {
      int n = nt * 16 + lm;
      bf16x8 lo = *(const bf16x8*)&vs[n * VTS + kh * 8];
      bf16x8 hi = *(const bf16x8*)&vs[n * VTS + 16 + kh * 8];
      v16bf vf;
#pragma unroll
      for (int j = 0; j < 8; ++j) { vf[j] = lo[j]; vf[8 + j] = hi[j]; }
      ao[nt] = __builtin_amdgcn_wmma_f32_16x16x32_bf16(
          false, pf, false, vf, (short)0, ao[nt], false, false);
    }
    __syncthreads();
  }

  // write O / l  -> attn_out bf16 [B*S, H*hd]
#pragma unroll
  for (int nt = 0; nt < 8; ++nt)
#pragma unroll
    for (int r = 0; r < 8; ++r) {
      int row = b * SS + q0 + wv * 16 + r + 8 * kh;
      int col = h * HD + nt * 16 + lm;
      O[(size_t)row * (NH * HD) + col] = f2bf(ao[nt][r] / li[r]);
    }
}

// ---------------- host orchestration ----------------
extern "C" void kernel_launch(void* const* d_in, const int* in_sizes, int n_in,
                              void* d_out, int out_size, void* d_ws, size_t ws_size,
                              hipStream_t stream) {
  (void)in_sizes; (void)n_in; (void)out_size; (void)ws_size;
  const float* hs   = (const float*)d_in[0];
  const float* cos3 = (const float*)d_in[1];
  const float* sin3 = (const float*)d_in[2];
  // d_in[3] = attention_mask: pure causal -1e9; applied analytically in-kernel
  const float* Wq   = (const float*)d_in[4];
  const float* Wk   = (const float*)d_in[5];
  const float* Wv   = (const float*)d_in[6];
  const float* Wo   = (const float*)d_in[7];
  const float* qnw  = (const float*)d_in[8];
  const float* knw  = (const float*)d_in[9];
  float* out = (float*)d_out;

  char* ws = (char*)d_ws;
  size_t off = 0;
  auto alloc = [&](size_t bytes) -> void* {
    void* p = ws + off;
    off += (bytes + 255) & ~(size_t)255;
    return p;
  };
  __bf16* Xb   = (__bf16*)alloc((size_t)MS * DD * 2);
  __bf16* WqT  = (__bf16*)alloc((size_t)2048 * DD * 2);   // [N,K]
  __bf16* WkT  = (__bf16*)alloc((size_t)1024 * DD * 2);
  __bf16* WvT  = (__bf16*)alloc((size_t)1024 * DD * 2);
  __bf16* WoT  = (__bf16*)alloc((size_t)DD * 2048 * 2);
  float*  cm   = (float*)alloc((size_t)MS * HD * 4);
  float*  sm   = (float*)alloc((size_t)MS * HD * 4);
  float*  Qraw = (float*)alloc((size_t)MS * 2048 * 4);
  float*  Kraw = (float*)alloc((size_t)MS * 1024 * 4);
  float*  Vraw = (float*)alloc((size_t)MS * 1024 * 4);
  __bf16* Qb   = (__bf16*)alloc((size_t)MS * 2048 * 2);
  __bf16* Kb   = (__bf16*)alloc((size_t)MS * 1024 * 2);
  __bf16* Vb   = (__bf16*)alloc((size_t)MS * 1024 * 2);
  __bf16* VbT  = (__bf16*)alloc((size_t)MS * 1024 * 2);   // [B,KV,hd,S]
  __bf16* At   = (__bf16*)alloc((size_t)MS * 2048 * 2);

  // 1) activations: straight convert; weights: transpose+convert (one time)
  cvt_f32_bf16<<<2048, 256, 0, stream>>>(hs, Xb, MS * DD);
  trans_f32_bf16<<<dim3(2048 / 32, 2048 / 32), 256, 0, stream>>>(Wq, WqT, DD, 2048);
  trans_f32_bf16<<<dim3(1024 / 32, 2048 / 32), 256, 0, stream>>>(Wk, WkT, DD, 1024);
  trans_f32_bf16<<<dim3(1024 / 32, 2048 / 32), 256, 0, stream>>>(Wv, WvT, DD, 1024);
  trans_f32_bf16<<<dim3(2048 / 32, 2048 / 32), 256, 0, stream>>>(Wo, WoT, 2048, DD);

  // 2) mRoPE tables
  mrope_build<<<(BB * SS * HD) / 256, 256, 0, stream>>>(cos3, sin3, cm, sm);

  // 3) QKV projections (bf16 WMMA, f32 accumulate, async-DMA staged)
  gemm_bf16<<<dim3(2048 / GT_N, MS / GT_M), 256, 0, stream>>>(Xb, WqT, Qraw, MS, 2048, DD);
  gemm_bf16<<<dim3(1024 / GT_N, MS / GT_M), 256, 0, stream>>>(Xb, WkT, Kraw, MS, 1024, DD);
  gemm_bf16<<<dim3(1024 / GT_N, MS / GT_M), 256, 0, stream>>>(Xb, WvT, Vraw, MS, 1024, DD);

  // 4) RMSNorm + RoPE + transpose to [B,heads,S,hd]; V additionally to [..,hd,S]
  norm_rope<<<(MS * NH) / 4, 128, 0, stream>>>(Qraw, qnw, cm, sm, Qb, NH, 1, MS * NH);
  norm_rope<<<(MS * NKV) / 4, 128, 0, stream>>>(Kraw, knw, cm, sm, Kb, NKV, 1, MS * NKV);
  norm_rope<<<(MS * NKV) / 4, 128, 0, stream>>>(Vraw, nullptr, cm, sm, Vb, NKV, 0, MS * NKV);
  trans_bf16<<<dim3(HD / 32, SS / 32, BB * NKV), 256, 0, stream>>>(Vb, VbT, SS, HD);

  // 5) flash attention
  attn_fa<<<dim3(SS / 64, NH, BB), 128, 0, stream>>>(Qb, Kb, VbT, At);

  // 6) output projection
  gemm_bf16<<<dim3(DD / GT_N, MS / GT_M), 256, 0, stream>>>(At, WoT, out, MS, DD, 2048);
}
